// MPBlock_43087111913969
// MI455X (gfx1250) — matrix operand order repaired and checked by reading, not compiled, not run on hardware
//
#include <hip/hip_runtime.h>
#include <cstddef>

// ---------------------------------------------------------------------------
// MI455X / gfx1250 implementation of the token-merging transformer block.
// GEMM-class math via v_wmma_f32_16x16x32_bf16 (wave32); GEMM LDS staging via
// the Tensor Data Mover (tensor_load_to_lds + s_wait_tensorcnt).
// ---------------------------------------------------------------------------

#define Bdim 16
#define Ndim 1024
#define Cdim 768
#define Hdim 12
#define DHdim 64
#define HIDdim 3072
#define Mrows (Bdim * Ndim)          // 16384

typedef __attribute__((ext_vector_type(16))) __bf16          v16bf;
typedef __attribute__((ext_vector_type(8)))  float           v8f;
typedef __attribute__((ext_vector_type(16))) unsigned short  u16x16;
typedef __attribute__((ext_vector_type(8)))  unsigned short  u16x8;
typedef __attribute__((ext_vector_type(4)))  unsigned int    u32x4;
typedef __attribute__((ext_vector_type(8)))  int             i32x8;

// ----- bf16 helpers (storage type = unsigned short) -------------------------
__device__ __forceinline__ unsigned short f2bf(float f) {
    unsigned int u = __builtin_bit_cast(unsigned int, f);
    u += 0x7FFFu + ((u >> 16) & 1u);       // round-to-nearest-even
    return (unsigned short)(u >> 16);
}
__device__ __forceinline__ float bf2f(unsigned short s) {
    unsigned int u = ((unsigned int)s) << 16;
    return __builtin_bit_cast(float, u);
}

// A-matrix (16x32 bf16) element K index for v16bf slot i, lane-half kh.
// ISA 7.12.2: lanes 0-15: V0..3 => K0..7, V4..7 => K16..23 ; lanes 16-31: +8.
// Equivalently: frag = concat(run of 8 at K=kh*8, run of 8 at K=16+kh*8).
__device__ __forceinline__ int amapK(int i, int kh) {
    int v = i >> 1;
    return ((v & 4) << 2) | (kh << 3) | ((v & 3) << 1) | (i & 1);
}

// Build a 16-element fragment from two contiguous 8-element runs (b128 loads).
__device__ __forceinline__ u16x16 frag_2runs(const unsigned short* r0,
                                             const unsigned short* r1) {
    u16x8 a = *(const u16x8*)r0;
    u16x8 b = *(const u16x8*)r1;
    return __builtin_shufflevector(a, b, 0, 1, 2, 3, 4, 5, 6, 7,
                                         8, 9, 10, 11, 12, 13, 14, 15);
}

__device__ __forceinline__ v8f wmma_bf16(const u16x16& a, const u16x16& b, v8f c) {
    v16bf av = __builtin_bit_cast(v16bf, a);
    v16bf bv = __builtin_bit_cast(v16bf, b);
    return __builtin_amdgcn_wmma_f32_16x16x32_bf16(false, av, false, bv,
                                                   (short)0, c, false, false);
}

// ---------------------------------------------------------------------------
// Tensor Data Mover: DMA a 2D bf16 tile (tileW x tileH rows) from global to
// LDS. D# built per ISA 8.3/8.4: group0 = {count, lds_addr, global_addr,
// type=2}; group1 = {data_size=2B, tensor dims, tile dims, dim0 stride}.
// Tensor dims == tile dims so no OOB path; rowStride = matrix pitch (elems).
// ---------------------------------------------------------------------------
__device__ __forceinline__ void tdm_load_2d(unsigned ldsOff, const void* gaddr,
                                            unsigned tileW, unsigned tileH,
                                            unsigned long long rowStride) {
    unsigned long long ga = (unsigned long long)gaddr;
    u32x4 g0;
    g0[0] = 1u;                                        // count=1, no gather
    g0[1] = ldsOff;                                    // lds_addr (bytes)
    g0[2] = (unsigned)(ga & 0xFFFFFFFFu);              // global_addr[31:0]
    g0[3] = (unsigned)((ga >> 32) & 0x01FFFFFFu)       // global_addr[56:32]
          | (2u << 30);                                // type = 2 ("image")
    i32x8 g1;
    g1[0] = (int)(1u << 16);                           // data_size=1 -> 2 bytes
    g1[1] = (int)((tileW & 0xFFFFu) << 16);            // tensor_dim0[15:0]
    g1[2] = (int)((tileW >> 16) | ((tileH & 0xFFFFu) << 16));  // d0 hi | d1 lo
    g1[3] = (int)((tileH >> 16) | (tileW << 16));      // d1 hi | tile_dim0
    g1[4] = (int)(tileH & 0xFFFFu);                    // tile_dim1 (dim2 = 0)
    g1[5] = (int)(unsigned)(rowStride & 0xFFFFFFFFull);          // d0 stride lo
    g1[6] = (int)(unsigned)((rowStride >> 32) & 0xFFFFull);      // d0 stride hi
    g1[7] = 0;
    u32x4 gz = {0u, 0u, 0u, 0u};                       // groups 2/3 unused (2D)
    asm volatile("tensor_load_to_lds %0, %1, %2, %3"
                 :: "s"(g0), "s"(g1), "s"(gz), "s"(gz)
                 : "memory");
}

// LDS byte offset of a __shared__ object: LDS aperture maps addr[31:0] -> LDS
// offset (ISA 10.2), so the low 32 bits of the generic pointer are the offset.
__device__ __forceinline__ unsigned lds_offset(const void* p) {
    return (unsigned)(unsigned long long)p;
}

// ---------------------------------------------------------------------------
// f32 -> bf16 convert (weights)
// ---------------------------------------------------------------------------
__global__ void f2bf_kernel(const float* __restrict__ in,
                            unsigned short* __restrict__ out, int n) {
    int i = blockIdx.x * blockDim.x + threadIdx.x;
    if (i < n) out[i] = f2bf(in[i]);
}

// ---------------------------------------------------------------------------
// Row LayerNorm (C=768), 256 threads/row, output bf16
// ---------------------------------------------------------------------------
__global__ __launch_bounds__(256)
void ln_kernel(const float* __restrict__ x, const float* __restrict__ w,
               const float* __restrict__ bsh, unsigned short* __restrict__ out) {
    __shared__ float red[16];
    int row = blockIdx.x;
    int t = threadIdx.x;
    const float* xr = x + (size_t)row * Cdim;
    float v0 = xr[t], v1 = xr[t + 256], v2 = xr[t + 512];
    float s = v0 + v1 + v2;
    float sq = v0 * v0 + v1 * v1 + v2 * v2;
    for (int m = 16; m >= 1; m >>= 1) { s += __shfl_xor(s, m); sq += __shfl_xor(sq, m); }
    int wave = t >> 5, lane = t & 31;
    if (lane == 0) { red[wave] = s; red[8 + wave] = sq; }
    __syncthreads();
    if (t == 0) {
        float S = 0.f, Q = 0.f;
        for (int i = 0; i < 8; ++i) { S += red[i]; Q += red[8 + i]; }
        float mean = S * (1.0f / Cdim);
        float var = Q * (1.0f / Cdim) - mean * mean;
        red[0] = mean;
        red[1] = rsqrtf(var + 1e-5f);
    }
    __syncthreads();
    float mean = red[0], inv = red[1];
    unsigned short* o = out + (size_t)row * Cdim;
    o[t]       = f2bf((v0 - mean) * inv * w[t]       + bsh[t]);
    o[t + 256] = f2bf((v1 - mean) * inv * w[t + 256] + bsh[t + 256]);
    o[t + 512] = f2bf((v2 - mean) * inv * w[t + 512] + bsh[t + 512]);
}

// ---------------------------------------------------------------------------
// Generic bf16 WMMA GEMM: D = A(MxK) @ B(KxN) (row-major), tile 128x128.
// 256 threads = 8 waves; wave (r,c) of 4x2 grid owns a 32x64 sub-tile.
// A/B tiles staged to LDS by the Tensor Data Mover (wave 0 issues, waits on
// TENSORcnt, workgroup barrier releases the consumers).
// Epilogues: 0=QKV scatter(bf16), 2=bias+GELU(bf16), 3=resid+bias(f32),
//            4=add bias+acc into existing f32 output.
// ---------------------------------------------------------------------------
template <int EPI>
__global__ __launch_bounds__(256)
void gemm_kernel(const unsigned short* __restrict__ A,
                 const unsigned short* __restrict__ Bw,
                 int M, int Nn, int K,
                 const float* __restrict__ bias,
                 const float* __restrict__ resid,
                 float* __restrict__ outF,
                 unsigned short* __restrict__ outBf,
                 unsigned short* __restrict__ qkv0,
                 int ldC) {
    __shared__ __align__(16) unsigned short As[128 * 32];
    __shared__ __align__(16) unsigned short Bs[32 * 128];
    const int t = threadIdx.x;
    const int lane = t & 31, wave = t >> 5;
    const int kh = lane >> 4, lm = lane & 15;
    const int wM = wave >> 1, wN = wave & 1;
    const int bM = blockIdx.y * 128, bN = blockIdx.x * 128;
    const unsigned asOff = lds_offset(As);
    const unsigned bsOff = lds_offset(Bs);

    v8f zero = {0.f, 0.f, 0.f, 0.f, 0.f, 0.f, 0.f, 0.f};
    v8f acc[2][4];
    for (int mt = 0; mt < 2; ++mt)
        for (int nt = 0; nt < 4; ++nt) acc[mt][nt] = zero;

    for (int k0 = 0; k0 < K; k0 += 32) {
        __syncthreads();                       // all waves done reading tiles
        if (wave == 0) {                       // TDM: async DMA both tiles
            tdm_load_2d(asOff, &A[(size_t)bM * K + k0], 32u, 128u,
                        (unsigned long long)K);
            tdm_load_2d(bsOff, &Bw[(size_t)k0 * Nn + bN], 128u, 32u,
                        (unsigned long long)Nn);
            __builtin_amdgcn_s_wait_tensorcnt(0);
        }
        __syncthreads();

        u16x16 ua[2];
#pragma unroll
        for (int mt = 0; mt < 2; ++mt) {       // A frag = two b128 LDS loads
            int row = wM * 32 + mt * 16 + lm;
            ua[mt] = frag_2runs(&As[row * 32 + kh * 16],
                                &As[row * 32 + kh * 16 + 8]);
        }
#pragma unroll
        for (int nt = 0; nt < 4; ++nt) {
            u16x16 ub;
            int col = wN * 64 + nt * 16 + lm;
#pragma unroll
            for (int i = 0; i < 16; ++i) ub[i] = Bs[(kh * 16 + i) * 128 + col];
#pragma unroll
            for (int mt = 0; mt < 2; ++mt)
                acc[mt][nt] = wmma_bf16(ua[mt], ub, acc[mt][nt]);
        }
    }

#pragma unroll
    for (int mt = 0; mt < 2; ++mt)
#pragma unroll
        for (int nt = 0; nt < 4; ++nt)
#pragma unroll
            for (int r = 0; r < 8; ++r) {
                int gr = bM + wM * 32 + mt * 16 + kh * 8 + r;   // C: M = r + 8*kh
                int gc = bN + wN * 64 + nt * 16 + lm;
                float val = acc[mt][nt][r];
                if constexpr (EPI == 0) {                        // QKV scatter
                    int which = gc / Cdim;
                    int rem = gc - which * Cdim;
                    int hh = rem >> 6, d = rem & 63;
                    int b = gr >> 10, tok = gr & 1023;
                    size_t stride = (size_t)Bdim * Hdim * Ndim * DHdim;
                    qkv0[(size_t)which * stride +
                         ((size_t)(b * Hdim + hh) * Ndim + tok) * DHdim + d] = f2bf(val);
                } else if constexpr (EPI == 2) {                 // bias + GELU -> bf16
                    float xg = val + bias[gc];
                    float g = 0.5f * xg * (1.0f + erff(xg * 0.70710678118654752f));
                    outBf[(size_t)gr * ldC + gc] = f2bf(g);
                } else if constexpr (EPI == 3) {                 // residual + bias -> f32
                    outF[(size_t)gr * ldC + gc] =
                        resid[(size_t)gr * ldC + gc] + val + bias[gc];
                } else {                                         // EPI==4: out += acc+bias
                    outF[(size_t)gr * ldC + gc] += val + bias[gc];
                }
            }
}

// ---------------------------------------------------------------------------
// Attention: one 128-thread WG per (b, h, 16-query tile). Scores (16x1024 f32)
// live in 64KB LDS; softmax with wave-shuffle reductions; P.V via WMMA.
// q/k fragments are vectorized b128 global loads (two 8xbf16 runs each).
// ---------------------------------------------------------------------------
__global__ __launch_bounds__(128)
void attn_kernel(const unsigned short* __restrict__ q,
                 const unsigned short* __restrict__ k,
                 const unsigned short* __restrict__ v,
                 unsigned short* __restrict__ xo,
                 float* __restrict__ imp_part) {
    __shared__ float sc[16 * 1024];                 // exactly 64 KB
    int wg = blockIdx.x;
    int qt = wg & 63;
    int h  = (wg >> 6) % Hdim;
    int b  = wg / (64 * Hdim);
    int t = threadIdx.x;
    int wave = t >> 5, lane = t & 31;
    int kh = lane >> 4, lm = lane & 15;
    size_t headBase = ((size_t)(b * Hdim + h)) * Ndim * DHdim;
    int q0 = qt * 16;

    // Q fragments (16x64, two K=32 slices), shared by all key tiles.
    u16x16 uq[2];
#pragma unroll
    for (int s = 0; s < 2; ++s) {
        const unsigned short* qr = q + headBase + (size_t)(q0 + lm) * DHdim + s * 32;
        uq[s] = frag_2runs(&qr[kh * 16], &qr[kh * 16 + 8]);
    }

    v8f zero = {0.f, 0.f, 0.f, 0.f, 0.f, 0.f, 0.f, 0.f};
    const float scale = 0.125f;                     // 64^-0.5
    for (int kt = wave; kt < 64; kt += 4) {         // 16-key tiles across 4 waves
        v8f a = zero;
#pragma unroll
        for (int s = 0; s < 2; ++s) {               // B[kk,key]: kk = s*32+kh*16+i
            const unsigned short* kr =
                k + headBase + (size_t)(kt * 16 + lm) * DHdim + s * 32 + kh * 16;
            u16x16 ub = frag_2runs(kr, kr + 8);
            a = wmma_bf16(uq[s], ub, a);
        }
#pragma unroll
        for (int r = 0; r < 8; ++r)
            sc[(kh * 8 + r) * 1024 + kt * 16 + lm] = a[r] * scale;
    }
    __syncthreads();

    // softmax: wave w owns rows 4w..4w+3; 8 lanes x 128 keys per row.
    {
        int row = wave * 4 + (lane >> 3);
        float* srow = &sc[row * 1024 + (lane & 7) * 128];
        float mx = -3.4e38f;
        for (int c = 0; c < 128; ++c) mx = fmaxf(mx, srow[c]);
        mx = fmaxf(mx, __shfl_xor(mx, 1));
        mx = fmaxf(mx, __shfl_xor(mx, 2));
        mx = fmaxf(mx, __shfl_xor(mx, 4));
        float sum = 0.f;
        for (int c = 0; c < 128; ++c) { float e = __expf(srow[c] - mx); srow[c] = e; sum += e; }
        sum += __shfl_xor(sum, 1);
        sum += __shfl_xor(sum, 2);
        sum += __shfl_xor(sum, 4);
        float inv = 1.0f / sum;
        for (int c = 0; c < 128; ++c) srow[c] *= inv;
    }
    __syncthreads();

    // importance partial: column sums over the 16 query rows (deterministic).
    for (int cc = t; cc < 1024; cc += 128) {
        float s = 0.f;
#pragma unroll
        for (int r = 0; r < 16; ++r) s += sc[r * 1024 + cc];
        imp_part[((size_t)b * (Hdim * 64) + (h * 64 + qt)) * 1024 + cc] = s;
    }

    // P.V : wave w owns output dims [16w, 16w+16)
    v8f acc = zero;
    int dcol = wave * 16 + lm;
    for (int kc = 0; kc < 32; ++kc) {
        u16x16 ua, ub;
#pragma unroll
        for (int i = 0; i < 16; ++i)
            ua[i] = f2bf(sc[lm * 1024 + kc * 32 + kh * 16 + i]);
#pragma unroll
        for (int i = 0; i < 16; ++i)
            ub[i] = v[headBase + (size_t)(kc * 32 + kh * 16 + i) * DHdim + dcol];
        acc = wmma_bf16(ua, ub, acc);
    }
#pragma unroll
    for (int r = 0; r < 8; ++r) {
        int qrow = q0 + kh * 8 + r;
        xo[((size_t)(b * Ndim + qrow)) * Cdim + h * DHdim + wave * 16 + lm] = f2bf(acc[r]);
    }
}

// importance[b,key] = (1/(H*N)) * sum over 768 partial rows (fixed order).
__global__ void imp_reduce_kernel(const float* __restrict__ part,
                                  float* __restrict__ imp) {
    int idx = blockIdx.x * blockDim.x + threadIdx.x;
    if (idx >= Bdim * Ndim) return;
    int b = idx >> 10, key = idx & 1023;
    const float* p = part + (size_t)b * (Hdim * 64) * 1024 + key;
    float s = 0.f;
    for (int j = 0; j < Hdim * 64; ++j) s += p[(size_t)j * 1024];
    imp[idx] = s * (1.0f / (Hdim * (float)Ndim));
}

// metric[b,t,:] = normalize(mean_h k[b,h,t,:])
__global__ void metric_kernel(const unsigned short* __restrict__ k,
                              float* __restrict__ metric) {
    int idx = blockIdx.x * blockDim.x + threadIdx.x;
    if (idx >= Bdim * Ndim) return;
    int b = idx >> 10, tok = idx & 1023;
    float m[DHdim];
#pragma unroll
    for (int d = 0; d < DHdim; ++d) m[d] = 0.f;
    for (int h = 0; h < Hdim; ++h) {
        size_t base = ((size_t)(b * Hdim + h) * Ndim + tok) * DHdim;
#pragma unroll
        for (int d = 0; d < DHdim; ++d) m[d] += bf2f(k[base + d]);
    }
    float ss = 0.f;
#pragma unroll
    for (int d = 0; d < DHdim; ++d) { m[d] *= (1.0f / Hdim); ss += m[d] * m[d]; }
    float inv = rsqrtf(ss);
    float* o = metric + (size_t)idx * DHdim;
#pragma unroll
    for (int d = 0; d < DHdim; ++d) o[d] = m[d] * inv;
}

// node_max[b,j] = max_jj metric_even[j] . metric_odd[jj]   (row 0 -> -inf)
__global__ void nodemax_kernel(const float* __restrict__ metric,
                               float* __restrict__ node_max) {
    int idx = blockIdx.x * blockDim.x + threadIdx.x;
    if (idx >= Bdim * 512) return;
    int b = idx >> 9, j = idx & 511;
    if (j == 0) { node_max[idx] = -__builtin_inff(); return; }
    const float* a = metric + ((size_t)b * Ndim + 2 * j) * DHdim;
    float ar[DHdim];
#pragma unroll
    for (int d = 0; d < DHdim; ++d) ar[d] = a[d];
    float best = -3.4e38f;
    for (int jj = 0; jj < 512; ++jj) {
        const float* bb = metric + ((size_t)b * Ndim + 2 * jj + 1) * DHdim;
        float dot = 0.f;
#pragma unroll
        for (int d = 0; d < DHdim; ++d) dot += ar[d] * bb[d];
        best = fmaxf(best, dot);
    }
    node_max[idx] = best;
}

// Stable double argsort: evens by node_max desc, then all 1024 by importance
// desc (token 0 forced +inf). perm[b,i] = original token index for slot i.
__global__ __launch_bounds__(256)
void build_perm_kernel(const float* __restrict__ node_max,
                       const float* __restrict__ imp,
                       int* __restrict__ perm) {
    __shared__ float nm[512];
    __shared__ int   eord[512];
    __shared__ float midv[1024];
    __shared__ int   midt[1024];
    int b = blockIdx.x, t = threadIdx.x;
    for (int j = t; j < 512; j += 256) nm[j] = node_max[b * 512 + j];
    __syncthreads();
    for (int j = t; j < 512; j += 256) {             // stable rank (descending)
        float vj = nm[j];
        int pos = 0;
        for (int i = 0; i < 512; ++i) {
            float vi = nm[i];
            pos += (vi > vj) || (vi == vj && i < j);
        }
        eord[pos] = j;
    }
    __syncthreads();
    for (int i = t; i < 1024; i += 256) {            // concat(permuted evens, odds)
        int tok = (i < 512) ? 2 * eord[i] : 2 * (i - 512) + 1;
        midv[i] = (tok == 0) ? __builtin_inff() : imp[b * Ndim + tok];
        midt[i] = tok;
    }
    __syncthreads();
    for (int i = t; i < 1024; i += 256) {            // stable rank by importance
        float vj = midv[i];
        int pos = 0;
        for (int jj = 0; jj < 1024; ++jj) {
            float vv = midv[jj];
            pos += (vv > vj) || (vv == vj && jj < i);
        }
        perm[b * Ndim + pos] = midt[i];
    }
}

// Gather rows by perm into d_out; size output = 1.0
__global__ __launch_bounds__(256)
void permute_kernel(const float* __restrict__ x1, const int* __restrict__ perm,
                    float* __restrict__ out_x, float* __restrict__ out_sz) {
    int row = blockIdx.x;                             // b*1024 + i
    int b = row >> 10;
    int src = perm[row];
    const float* s = x1 + ((size_t)b * Ndim + src) * Cdim;
    float* d = out_x + (size_t)row * Cdim;
    for (int c = threadIdx.x; c < Cdim; c += 256) d[c] = s[c];
    if (threadIdx.x == 0) out_sz[row] = 1.0f;
}

// ---------------------------------------------------------------------------
extern "C" void kernel_launch(void* const* d_in, const int* in_sizes, int n_in,
                              void* d_out, int out_size, void* d_ws, size_t ws_size,
                              hipStream_t stream) {
    (void)in_sizes; (void)n_in; (void)out_size; (void)ws_size;
    const float* x    = (const float*)d_in[0];
    const float* n1w  = (const float*)d_in[2];
    const float* n1b  = (const float*)d_in[3];
    const float* qkvw = (const float*)d_in[4];
    const float* pw   = (const float*)d_in[5];
    const float* pb   = (const float*)d_in[6];
    const float* n2w  = (const float*)d_in[7];
    const float* n2b  = (const float*)d_in[8];
    const float* f1w  = (const float*)d_in[9];
    const float* f1b  = (const float*)d_in[10];
    const float* f2w  = (const float*)d_in[11];
    const float* f2b  = (const float*)d_in[12];

    constexpr size_t QKVS   = (size_t)Bdim * Hdim * Ndim * DHdim;   // per-matrix elems
    constexpr size_t SZ_ACT = (size_t)Mrows * Cdim;

    char* ws = (char*)d_ws;
    size_t off = 0;
    auto alloc = [&](size_t bytes) {
        size_t o = off;
        off = (off + bytes + 255) & ~(size_t)255;
        return o;
    };
    unsigned short* qkvw_bf = (unsigned short*)(ws + alloc((size_t)Cdim * 3 * Cdim * 2));
    unsigned short* projw_bf= (unsigned short*)(ws + alloc((size_t)Cdim * Cdim * 2));
    unsigned short* fc1w_bf = (unsigned short*)(ws + alloc((size_t)Cdim * HIDdim * 2));
    unsigned short* fc2w_bf = (unsigned short*)(ws + alloc((size_t)HIDdim * Cdim * 2));
    unsigned short* xa_bf   = (unsigned short*)(ws + alloc(SZ_ACT * 2));     // also xm
    size_t o_qkv = alloc(3 * QKVS * 2);                                      // q,k,v
    unsigned short* q_bf = (unsigned short*)(ws + o_qkv);
    unsigned short* k_bf = q_bf + QKVS;
    unsigned short* v_bf = q_bf + 2 * QKVS;
    float* x1 = (float*)(ws + o_qkv);          // reuse: x1 (50MB) after k/v done
    unsigned short* xo_bf = (unsigned short*)(ws + alloc(SZ_ACT * 2));
    size_t o_h1 = alloc((size_t)Mrows * HIDdim * 2);
    unsigned short* h1_bf = (unsigned short*)(ws + o_h1);
    float* imp_part = (float*)(ws + o_h1);     // reuse: partials (50MB) before h1
    float* imp    = (float*)(ws + alloc((size_t)Bdim * Ndim * 4));
    float* metric = (float*)(ws + alloc((size_t)Bdim * Ndim * DHdim * 4));
    float* nmax   = (float*)(ws + alloc((size_t)Bdim * 512 * 4));
    int*   perm   = (int*)  (ws + alloc((size_t)Bdim * Ndim * 4));

    float* out_x      = (float*)d_out;
    float* out_size_p = out_x + (size_t)Mrows * Cdim;

    // 1) weights -> bf16
    auto conv = [&](const float* src, unsigned short* dst, size_t n) {
        f2bf_kernel<<<(unsigned)((n + 255) / 256), 256, 0, stream>>>(src, dst, (int)n);
    };
    conv(qkvw, qkvw_bf, (size_t)Cdim * 3 * Cdim);
    conv(pw,   projw_bf, (size_t)Cdim * Cdim);
    conv(f1w,  fc1w_bf, (size_t)Cdim * HIDdim);
    conv(f2w,  fc2w_bf, (size_t)HIDdim * Cdim);

    // 2) LN1 -> bf16
    ln_kernel<<<Mrows, 256, 0, stream>>>(x, n1w, n1b, xa_bf);

    // 3) QKV GEMM (16384 x 2304, K=768), scatter to (B,H,N,DH) bf16
    gemm_kernel<0><<<dim3(3 * Cdim / 128, Mrows / 128), 256, 0, stream>>>(
        xa_bf, qkvw_bf, Mrows, 3 * Cdim, Cdim,
        nullptr, nullptr, nullptr, nullptr, q_bf, 0);

    // 4) attention (WMMA scores + softmax + WMMA PV), importance partials
    attn_kernel<<<Bdim * Hdim * 64, 128, 0, stream>>>(q_bf, k_bf, v_bf, xo_bf, imp_part);

    // 5) importance reduce (deterministic), metric, node_max
    imp_reduce_kernel<<<Mrows / 256, 256, 0, stream>>>(imp_part, imp);
    metric_kernel<<<Mrows / 256, 256, 0, stream>>>(k_bf, metric);
    nodemax_kernel<<<(Bdim * 512) / 256, 256, 0, stream>>>(metric, nmax);

    // 6) proj GEMM + residual: x1 = x + xo@proj_w + proj_b   (x1 reuses qkv mem)
    gemm_kernel<3><<<dim3(Cdim / 128, Mrows / 128), 256, 0, stream>>>(
        xo_bf, projw_bf, Mrows, Cdim, Cdim,
        pb, x, x1, nullptr, nullptr, Cdim);

    // 7) permutation (stable argsorts) + gather into d_out, size = 1
    build_perm_kernel<<<Bdim, 256, 0, stream>>>(nmax, imp, perm);
    permute_kernel<<<Mrows, 256, 0, stream>>>(x1, perm, out_x, out_size_p);

    // 8) MLP: LN2 -> FC1+GELU -> FC2 added into d_out
    ln_kernel<<<Mrows, 256, 0, stream>>>(out_x, n2w, n2b, xa_bf);
    gemm_kernel<2><<<dim3(HIDdim / 128, Mrows / 128), 256, 0, stream>>>(
        xa_bf, fc1w_bf, Mrows, HIDdim, Cdim,
        f1b, nullptr, nullptr, h1_bf, nullptr, HIDdim);
    gemm_kernel<4><<<dim3(Cdim / 128, Mrows / 128), 256, 0, stream>>>(
        h1_bf, fc2w_bf, Mrows, Cdim, HIDdim,
        f2b, nullptr, out_x, nullptr, nullptr, Cdim);
}
